// Vision_Transformer_34763465294048
// MI455X (gfx1250) — matrix operand order repaired and checked by reading, not compiled, not run on hardware
//
#include <hip/hip_runtime.h>
#include <hip/hip_bf16.h>

// ---------------------------------------------------------------------------
// ViT block for MI455X (gfx1250): all GEMM stages via v_wmma_f32_16x16x32_f16
// wave32, 4 waves per workgroup, each wave computes a 16x64 f32 tile.
// Round 2: all operands pre-converted to f16 once (k_cvt) so the hot GEMM
// loops are pure  global_load_b128 -> v_wmma  with no per-iteration cvt VALU.
// ---------------------------------------------------------------------------

typedef __attribute__((ext_vector_type(16))) _Float16 v16h;
typedef __attribute__((ext_vector_type(8)))  _Float16 v8h;
typedef __attribute__((ext_vector_type(8)))  float    v8f;
typedef __attribute__((ext_vector_type(4)))  float    v4f;

#define HROWS   18432   // B*S = 32*576
#define EMB     512
#define S_LEN   576
#define HEADS   8
#define HDIM    64
#define KDIM    768     // CIN*PATCH*PATCH
#define XN      14155776 // 32*3*384*384

static __device__ __forceinline__ v8f wmma_f16(v16h a, v16h b, v8f c) {
  return __builtin_amdgcn_wmma_f32_16x16x32_f16(false, a, false, b, (short)0, c,
                                                false, false);
}

// B-fragment (32x16, f16): 16 contiguous K values per lane (32B load).
static __device__ __forceinline__ v16h ldB_h16(const _Float16* p) {
  return *(const v16h*)p;
}

// A-fragment (16x32, f16). Layout: halves 0..15 <-> K = hi*16 + j.
// base points at the row's 32-K block.
static __device__ __forceinline__ v16h ldA_h16(const _Float16* base, int hi) {
  v8h lo = *(const v8h*)(base + hi * 16);
  v8h up = *(const v8h*)(base + hi * 16 + 8);
  v16h r;
#pragma unroll
  for (int j = 0; j < 8; ++j) { r[j] = lo[j]; r[8 + j] = up[j]; }
  return r;
}

// A-fragment whose halves 8..15 lie `rstride` halves past halves 0..7 (K+8 at the call).
static __device__ __forceinline__ v16h ldA_h16_str(const _Float16* base, int hi,
                                                   int rstride) {
  v8h lo = *(const v8h*)(base + hi * 8);
  v8h up = *(const v8h*)(base + rstride + hi * 8);
  v16h r;
#pragma unroll
  for (int j = 0; j < 8; ++j) { r[j] = lo[j]; r[8 + j] = up[j]; }
  return r;
}

// ---------------------------------------------------------------------------
// Stage 0: bulk f32 -> f16 conversion (8 elements / thread, b128 in, b128 out)
// ---------------------------------------------------------------------------
__global__ void k_cvt(const float* __restrict__ src, _Float16* __restrict__ dst,
                      int n) {
  const int i = (blockIdx.x * 256 + threadIdx.x) * 8;
  if (i >= n) return;
  const v4f* s = (const v4f*)(src + i);
  v4f a = s[0], b = s[1];
  v8h r;
#pragma unroll
  for (int j = 0; j < 4; ++j) {
    r[j]     = (_Float16)a[j];
    r[4 + j] = (_Float16)b[j];
  }
  *(v8h*)(dst + i) = r;
}

// ---------------------------------------------------------------------------
// Stage 1: patch embedding  h[row][e] = sum_k patch(row)[k] * conv_w[e][k] + b[e]
// grid (288, 8), block 128.  K = 768, 24 WMMA k-steps.  All-f16 operands.
// ---------------------------------------------------------------------------
__global__ void k_patch_embed(const _Float16* __restrict__ xh,
                              const _Float16* __restrict__ cw,
                              const float* __restrict__ bias,
                              _Float16* __restrict__ h) {
  const int lane = threadIdx.x & 31;
  const int wv   = threadIdx.x >> 5;
  const int hi   = lane >> 4;
  const int ln   = lane & 15;
  const int rowBase = blockIdx.x * 64 + wv * 16;
  const int row  = rowBase + ln;          // A-layout row (M = ln)
  const int e0   = blockIdx.y * 64;
  const int b    = row / S_LEN;
  const int s    = row - b * S_LEN;
  const int py   = s / 24, px = s - py * 24;
  const _Float16* xrow =
      xh + (((size_t)(b * 3) * 384 + py * 16) * 384 + px * 16);

  v8f acc[4] = {};
  for (int ks = 0; ks < 24; ++ks) {
    const int k0  = ks * 32;
    const int kA  = k0 + hi * 16;         // first K of this lane's low run
    const int c   = kA >> 8;
    const int rem = kA & 255;
    const int ph  = rem >> 4;
    const int pw  = rem & 15;
    const _Float16* ax = xrow + ((size_t)c * 384 + ph) * 384 + pw;
    v16h a = ldA_h16_str(ax, 0, 8);       // hi already folded into ph
    __builtin_prefetch(ax + 2 * 384, 0, 0);
#pragma unroll
    for (int cb = 0; cb < 4; ++cb) {
      const _Float16* wp = cw + (size_t)(e0 + cb * 16 + ln) * KDIM + k0 + hi * 16;
      acc[cb] = wmma_f16(a, ldB_h16(wp), acc[cb]);
    }
  }
#pragma unroll
  for (int cb = 0; cb < 4; ++cb) {
    const int e  = e0 + cb * 16 + ln;
    const float bv = bias[e];
#pragma unroll
    for (int g = 0; g < 8; ++g) {         // C-layout row M = g + 8*hi
      h[(size_t)(rowBase + g + hi * 8) * EMB + e] = (_Float16)(acc[cb][g] + bv);
    }
  }
}

// ---------------------------------------------------------------------------
// Stage 2: q/k/v head projections.  z = blockIdx.z selects weight (f16).
// q,k stored [bh][s][d]; v stored transposed [bh][d][s] for att*V B-frags.
// grid (288, 8, 3), block 128.
// ---------------------------------------------------------------------------
__global__ void k_qkv(const _Float16* __restrict__ h,
                      const _Float16* __restrict__ wqh,
                      const _Float16* __restrict__ wkh,
                      const _Float16* __restrict__ wvh,
                      _Float16* __restrict__ q,
                      _Float16* __restrict__ kk,
                      _Float16* __restrict__ vt) {
  const int lane = threadIdx.x & 31;
  const int wav  = threadIdx.x >> 5;
  const int hi   = lane >> 4;
  const int ln   = lane & 15;
  const int rowBase = blockIdx.x * 64 + wav * 16;
  const int head = blockIdx.y;
  const int z    = blockIdx.z;
  const _Float16* wm = (z == 0) ? wqh : (z == 1) ? wkh : wvh;
  const int row  = rowBase + ln;

  v8f acc[4] = {};
#pragma unroll
  for (int ks = 0; ks < 2; ++ks) {
    v16h a = ldA_h16(h + (size_t)row * EMB + head * HDIM + ks * 32, hi);
#pragma unroll
    for (int cb = 0; cb < 4; ++cb) {
      const _Float16* wp = wm + (size_t)(cb * 16 + ln) * HDIM + ks * 32 + hi * 16;
      acc[cb] = wmma_f16(a, ldB_h16(wp), acc[cb]);
    }
  }

  const int b   = rowBase / S_LEN;
  const int s0  = rowBase - b * S_LEN;
  const size_t bh = (size_t)(b * HEADS + head);
  if (z < 2) {
    _Float16* dst = (z == 0 ? q : kk) + bh * S_LEN * HDIM;
#pragma unroll
    for (int cb = 0; cb < 4; ++cb)
#pragma unroll
      for (int g = 0; g < 8; ++g)
        dst[(size_t)(s0 + g + hi * 8) * HDIM + cb * 16 + ln] =
            (_Float16)acc[cb][g];
  } else {
    _Float16* dst = vt + bh * HDIM * S_LEN;
#pragma unroll
    for (int cb = 0; cb < 4; ++cb)
#pragma unroll
      for (int g = 0; g < 8; ++g)
        dst[(size_t)(cb * 16 + ln) * S_LEN + s0 + g + hi * 8] =
            (_Float16)acc[cb][g];
  }
}

// ---------------------------------------------------------------------------
// Stage 3: attention per (b,head).  Each wave: 16 query rows, two-pass softmax.
// grid (256, 9), block 128.  LDS: 1KB/wave P-tile transpose (C-layout->A-layout).
// ---------------------------------------------------------------------------
__global__ void k_attn(const _Float16* __restrict__ q,
                       const _Float16* __restrict__ kk,
                       const _Float16* __restrict__ vt,
                       _Float16* __restrict__ attn) {
  __shared__ __align__(64) _Float16 Pt[4][16][32];
  const int lane = threadIdx.x & 31;
  const int wav  = threadIdx.x >> 5;
  const int hi   = lane >> 4;
  const int ln   = lane & 15;
  const int bh   = blockIdx.x;
  const int b    = bh >> 3;
  const int head = bh & 7;
  const int s0   = blockIdx.y * 64 + wav * 16;
  const _Float16* qp = q  + (size_t)bh * S_LEN * HDIM;
  const _Float16* kp = kk + (size_t)bh * S_LEN * HDIM;
  const _Float16* vp = vt + (size_t)bh * HDIM * S_LEN;
  const float scale = 0.125f;   // 1/sqrt(64)

  v16h aq0 = ldA_h16(qp + (size_t)(s0 + ln) * HDIM + 0,  hi);
  v16h aq1 = ldA_h16(qp + (size_t)(s0 + ln) * HDIM + 32, hi);

  // ---- pass 1: row max --------------------------------------------------
  float mrow[8];
#pragma unroll
  for (int g = 0; g < 8; ++g) mrow[g] = -1e30f;
  for (int kb = 0; kb < 18; ++kb) {
#pragma unroll
    for (int sub = 0; sub < 2; ++sub) {
      const int key0 = kb * 32 + sub * 16;
      v16h b0 = ldB_h16(kp + (size_t)(key0 + ln) * HDIM + hi * 16);
      v16h b1 = ldB_h16(kp + (size_t)(key0 + ln) * HDIM + 32 + hi * 16);
      v8f c = {};
      c = wmma_f16(aq0, b0, c);
      c = wmma_f16(aq1, b1, c);
#pragma unroll
      for (int g = 0; g < 8; ++g) mrow[g] = fmaxf(mrow[g], c[g] * scale);
    }
  }
#pragma unroll
  for (int g = 0; g < 8; ++g) {
    float v = mrow[g];
    v = fmaxf(v, __shfl_xor(v, 1));
    v = fmaxf(v, __shfl_xor(v, 2));
    v = fmaxf(v, __shfl_xor(v, 4));
    v = fmaxf(v, __shfl_xor(v, 8));
    mrow[g] = v;                          // row max for M = g + 8*hi
  }

  // ---- pass 2: exp / sum / P*V ------------------------------------------
  v8f o[4] = {};
  float srow[8] = {0.f, 0.f, 0.f, 0.f, 0.f, 0.f, 0.f, 0.f};
  for (int kb = 0; kb < 18; ++kb) {
#pragma unroll
    for (int sub = 0; sub < 2; ++sub) {
      const int key0 = kb * 32 + sub * 16;
      v16h b0 = ldB_h16(kp + (size_t)(key0 + ln) * HDIM + hi * 16);
      v16h b1 = ldB_h16(kp + (size_t)(key0 + ln) * HDIM + 32 + hi * 16);
      v8f c = {};
      c = wmma_f16(aq0, b0, c);
      c = wmma_f16(aq1, b1, c);
#pragma unroll
      for (int g = 0; g < 8; ++g) {
        float p = __expf(c[g] * scale - mrow[g]);
        srow[g] += p;
        Pt[wav][g + hi * 8][sub * 16 + ln] = (_Float16)p;
      }
    }
    __builtin_amdgcn_wave_barrier();
    asm volatile("s_wait_dscnt 0" ::: "memory");
    v16h pa = ldA_h16(&Pt[wav][ln][0], hi);
    __builtin_amdgcn_wave_barrier();
    asm volatile("" ::: "memory");
#pragma unroll
    for (int cb = 0; cb < 4; ++cb) {
      v16h bv = ldB_h16(vp + (size_t)(cb * 16 + ln) * S_LEN + kb * 32 + hi * 16);
      o[cb] = wmma_f16(pa, bv, o[cb]);
    }
  }
#pragma unroll
  for (int g = 0; g < 8; ++g) {
    float v = srow[g];
    v += __shfl_xor(v, 1);
    v += __shfl_xor(v, 2);
    v += __shfl_xor(v, 4);
    v += __shfl_xor(v, 8);
    srow[g] = 1.0f / v;
  }
#pragma unroll
  for (int cb = 0; cb < 4; ++cb)
#pragma unroll
    for (int g = 0; g < 8; ++g) {
      const int m = g + hi * 8;
      attn[(size_t)(b * S_LEN + s0 + m) * EMB + head * HDIM + cb * 16 + ln] =
          (_Float16)(o[cb][g] * srow[g]);
    }
}

// ---------------------------------------------------------------------------
// Stage 4: output projection  out = attn @ wo^T + bo   (f32 result)
// grid (288, 8), block 128.  K = 512, 16 WMMA k-steps.  f16 weights.
// ---------------------------------------------------------------------------
__global__ void k_proj(const _Float16* __restrict__ attn,
                       const _Float16* __restrict__ woh,
                       const float* __restrict__ bo,
                       float* __restrict__ out) {
  const int lane = threadIdx.x & 31;
  const int wav  = threadIdx.x >> 5;
  const int hi   = lane >> 4;
  const int ln   = lane & 15;
  const int rowBase = blockIdx.x * 64 + wav * 16;
  const int e0   = blockIdx.y * 64;

  v8f acc[4] = {};
  for (int ks = 0; ks < 16; ++ks) {
    const _Float16* ap = attn + (size_t)(rowBase + ln) * EMB + ks * 32;
    v16h a = ldA_h16(ap, hi);
    __builtin_prefetch(ap + 32, 0, 0);
#pragma unroll
    for (int cb = 0; cb < 4; ++cb) {
      const _Float16* wp = woh + (size_t)(e0 + cb * 16 + ln) * EMB + ks * 32 + hi * 16;
      acc[cb] = wmma_f16(a, ldB_h16(wp), acc[cb]);
    }
  }
#pragma unroll
  for (int cb = 0; cb < 4; ++cb) {
    const int e  = e0 + cb * 16 + ln;
    const float bb = bo[e];
#pragma unroll
    for (int g = 0; g < 8; ++g)
      out[(size_t)(rowBase + g + hi * 8) * EMB + e] = acc[cb][g] + bb;
  }
}

// ---------------------------------------------------------------------------
extern "C" void kernel_launch(void* const* d_in, const int* in_sizes, int n_in,
                              void* d_out, int out_size, void* d_ws, size_t ws_size,
                              hipStream_t stream) {
  const float* x      = (const float*)d_in[0];
  const float* conv_w = (const float*)d_in[1];
  const float* conv_b = (const float*)d_in[2];
  const float* wq     = (const float*)d_in[3];
  const float* wk     = (const float*)d_in[4];
  const float* wv     = (const float*)d_in[5];
  const float* wo     = (const float*)d_in[6];
  const float* bo     = (const float*)d_in[7];
  float* out = (float*)d_out;

  const size_t NB = (size_t)HROWS * EMB;   // 9,437,184 halves per buffer
  _Float16* ws   = (_Float16*)d_ws;
  _Float16* h    = ws;                     // aliased by attn (h dead after qkv)
  _Float16* q    = ws + NB;
  _Float16* kk   = ws + 2 * NB;
  _Float16* vt   = ws + 3 * NB;
  _Float16* xh   = ws + 4 * NB;            // f16 image          (28.3 MB)
  _Float16* cwh  = xh + (size_t)XN;        // f16 conv_w [512][768]
  _Float16* wqh  = cwh + (size_t)EMB * KDIM;
  _Float16* wkh  = wqh + HDIM * HDIM;
  _Float16* wvh  = wkh + HDIM * HDIM;
  _Float16* woh  = wvh + HDIM * HDIM;      // f16 wo [512][512]
  _Float16* attn = h;

  dim3 blk(128);
  dim3 cblk(256);
  // Stage 0: one-shot f32 -> f16 operand conversion.
  k_cvt<<<dim3(XN / (8 * 256)), cblk, 0, stream>>>(x, xh, XN);
  k_cvt<<<dim3((EMB * KDIM) / (8 * 256)), cblk, 0, stream>>>(conv_w, cwh,
                                                             EMB * KDIM);
  k_cvt<<<dim3(2), cblk, 0, stream>>>(wq, wqh, HDIM * HDIM);
  k_cvt<<<dim3(2), cblk, 0, stream>>>(wk, wkh, HDIM * HDIM);
  k_cvt<<<dim3(2), cblk, 0, stream>>>(wv, wvh, HDIM * HDIM);
  k_cvt<<<dim3((EMB * EMB) / (8 * 256)), cblk, 0, stream>>>(wo, woh, EMB * EMB);

  k_patch_embed<<<dim3(288, 8),    blk, 0, stream>>>(xh, cwh, conv_b, h);
  k_qkv        <<<dim3(288, 8, 3), blk, 0, stream>>>(h, wqh, wkh, wvh, q, kk, vt);
  k_attn       <<<dim3(256, 9),    blk, 0, stream>>>(q, kk, vt, attn);
  k_proj       <<<dim3(288, 8),    blk, 0, stream>>>(attn, woh, bo, out);
}
